// _WindowAttn_6588479832464
// MI455X (gfx1250) — matrix operand order repaired and checked by reading, not compiled, not run on hardware
//
#include <hip/hip_runtime.h>
#include <hip/hip_bf16.h>

// ---------------------------------------------------------------------------
// Window attention (Swin) for MI455X / gfx1250.
// bf16 WMMA (v_wmma_f32_16x16x32_bf16) everywhere; fp32 accumulate.
// ---------------------------------------------------------------------------

typedef __bf16 bf16_t;
typedef __attribute__((ext_vector_type(16))) __bf16 v16bf;
typedef __attribute__((ext_vector_type(8)))  __bf16 v8bf;
typedef __attribute__((ext_vector_type(4)))  __bf16 v4bf;
typedef __attribute__((ext_vector_type(8)))  float  v8f;
typedef __attribute__((ext_vector_type(4)))  float  v4f;

#define NHD   8      // heads
#define HD    32     // head dim
#define NWIN  49     // tokens per window (7x7)
#define CDIM  256    // channels
#define MTOT  (4096 * NWIN)   // 200704 rows
#define NPAD  64     // padded token count

union BF16Frag { v16bf v; v8bf h[2]; };

// A-matrix fragment (16x32 bf16, MxK). p = rowPtr + kbase + (lane>>4)*16.
// elements 0..15 = K[+0..+15] (contiguous).
__device__ __forceinline__ v16bf ld_a_frag(const bf16_t* p) {
    BF16Frag u;
    u.h[0] = *(const v8bf*)(p);
    u.h[1] = *(const v8bf*)(p + 8);
    return u.v;
}

// B-matrix fragment (32x16 bf16, KxN). Lane holds 16 CONTIGUOUS K values of
// its column: p = colPtr + kbase + (lane>>4)*16.
__device__ __forceinline__ v16bf ld_b_frag(const bf16_t* p) {
    BF16Frag u;
    u.h[0] = *(const v8bf*)(p);
    u.h[1] = *(const v8bf*)(p + 8);
    return u.v;
}

__device__ __forceinline__ v8f wmma_bf16(v16bf a, v16bf b, v8f c) {
    return __builtin_amdgcn_wmma_f32_16x16x32_bf16(false, a, false, b,
                                                   (short)0, c, false, false);
}

// ---------------------------------------------------------------------------
// Kernel 0: f32 -> bf16 conversion (vectorized, grid-stride).
// ---------------------------------------------------------------------------
__global__ void wa_cvt_bf16(const float* __restrict__ in,
                            bf16_t* __restrict__ out, int n4) {
    int i = blockIdx.x * blockDim.x + threadIdx.x;
    int stride = gridDim.x * blockDim.x;
    for (; i < n4; i += stride) {
        v4f f = *(const v4f*)(in + 4 * (size_t)i);
        v4bf o;
        #pragma unroll
        for (int j = 0; j < 4; ++j) o[j] = (bf16_t)f[j];
        *(v4bf*)(out + 4 * (size_t)i) = o;
    }
}

// ---------------------------------------------------------------------------
// Kernel 1: QKV GEMM. out[m][n] = sum_k x[m][k] * W[n][k] + bias[n]
// M=200704, K=256, N=768.  Grid (3136, 12), block 128 (4 waves).
// Epilogue scatters into q [b,h,49,32] (scaled), k [b,h,49,32], vT [b,h,32,64].
// ---------------------------------------------------------------------------
__global__ __launch_bounds__(128)
void wa_qkv_gemm(const bf16_t* __restrict__ xb, const bf16_t* __restrict__ wb,
                 const float* __restrict__ qkv_b,
                 bf16_t* __restrict__ qbuf, bf16_t* __restrict__ kbuf,
                 bf16_t* __restrict__ vtbuf) {
    const int lane   = threadIdx.x & 31;
    const int wave   = threadIdx.x >> 5;
    const int m_base = blockIdx.x * 64 + wave * 16;
    const int n_base = blockIdx.y * 64;
    const int l15    = lane & 15;
    const int hi     = lane >> 4;

    const bf16_t* aptr = xb + (size_t)(m_base + l15) * CDIM + hi * 16;
    const bf16_t* bptr = wb + (size_t)(n_base + l15) * CDIM + hi * 16;

    v8f c[4] = {};
    #pragma unroll
    for (int kk = 0; kk < 8; ++kk) {
        v16bf a = ld_a_frag(aptr + kk * 32);
        #pragma unroll
        for (int j = 0; j < 4; ++j) {
            v16bf b = ld_b_frag(bptr + (size_t)j * 16 * CDIM + kk * 32);
            c[j] = wmma_bf16(a, b, c[j]);
        }
    }

    const float scale = 0.17677669529663687f;  // 1/sqrt(32)
    #pragma unroll
    for (int j = 0; j < 4; ++j) {
        #pragma unroll
        for (int r = 0; r < 8; ++r) {
            int gm = m_base + r + 8 * hi;
            int n  = n_base + j * 16 + l15;
            float val = c[j][r] + qkv_b[n];
            int b_  = gm / NWIN;
            int key = gm - b_ * NWIN;
            if (n < CDIM) {                              // Q (pre-scaled)
                int h = n >> 5, d = n & 31;
                qbuf[(((size_t)b_ * NHD + h) * NWIN + key) * HD + d] =
                    (bf16_t)(val * scale);
            } else if (n < 2 * CDIM) {                   // K
                int nn = n - CDIM; int h = nn >> 5, d = nn & 31;
                kbuf[(((size_t)b_ * NHD + h) * NWIN + key) * HD + d] =
                    (bf16_t)val;
            } else {                                     // V, transposed
                int nn = n - 2 * CDIM; int h = nn >> 5, d = nn & 31;
                vtbuf[(((size_t)b_ * NHD + h) * HD + d) * NPAD + key] =
                    (bf16_t)val;
            }
        }
    }
}

// ---------------------------------------------------------------------------
// Kernel 2: per-(window, head) attention.  Grid (4096, 8), block 128.
// Wave w owns query rows 16w..16w+15 (N padded 49 -> 64).
// ---------------------------------------------------------------------------
__global__ __launch_bounds__(128)
void wa_attn(const bf16_t* __restrict__ qbuf, const bf16_t* __restrict__ kbuf,
             const bf16_t* __restrict__ vtbuf, const float* __restrict__ mask,
             const float* __restrict__ rpb, const int* __restrict__ rpi,
             bf16_t* __restrict__ abuf) {
    __shared__ float  SL[4][16][NPAD];   // scores (f32) per wave
    __shared__ bf16_t PL[4][16][NPAD];   // probs (bf16) per wave

    const int b    = blockIdx.x;
    const int h    = blockIdx.y;
    const int lane = threadIdx.x & 31;
    const int wave = threadIdx.x >> 5;
    const int l15  = lane & 15;
    const int hi   = lane >> 4;

    const bf16_t* qh = qbuf  + ((size_t)b * NHD + h) * NWIN * HD;
    const bf16_t* kh = kbuf  + ((size_t)b * NHD + h) * NWIN * HD;
    const bf16_t* vh = vtbuf + ((size_t)b * NHD + h) * HD * NPAD;

    // --- S = q * k^T  (16x64 strip, 4 WMMAs) ---
    int qr = wave * 16 + l15; if (qr > NWIN - 1) qr = NWIN - 1;
    v16bf qa = ld_a_frag(qh + (size_t)qr * HD + hi * 16);
    v8f s[4] = {};
    #pragma unroll
    for (int j = 0; j < 4; ++j) {
        int kr = j * 16 + l15; if (kr > NWIN - 1) kr = NWIN - 1;
        v16bf kb = ld_b_frag(kh + (size_t)kr * HD + hi * 16);
        s[j] = wmma_bf16(qa, kb, s[j]);
    }

    // --- bias + mask, stash f32 scores in LDS ---
    const float* mw = mask + (size_t)(b & 63) * (NWIN * NWIN);
    #pragma unroll
    for (int j = 0; j < 4; ++j) {
        #pragma unroll
        for (int r = 0; r < 8; ++r) {
            int rloc = r + 8 * hi;
            int row  = wave * 16 + rloc;
            int col  = j * 16 + l15;
            float v;
            if (col < NWIN) {
                int rr = row < NWIN ? row : NWIN - 1;
                v = s[j][r] + rpb[rpi[rr * NWIN + col] * NHD + h]
                            + mw[rr * NWIN + col];
            } else {
                v = -1e30f;                 // padded key columns
            }
            SL[wave][rloc][col] = v;
        }
    }
    __syncthreads();

    // --- softmax: 2 lanes per row, 32 cols each, pairwise shfl reduce ---
    {
        int row = lane >> 1;
        int cb  = (lane & 1) * 32;
        float mx = -1e30f;
        #pragma unroll
        for (int cc = 0; cc < 32; ++cc)
            mx = fmaxf(mx, SL[wave][row][cb + cc]);
        mx = fmaxf(mx, __shfl_xor(mx, 1, 32));
        float ev[32];
        float sum = 0.f;
        #pragma unroll
        for (int cc = 0; cc < 32; ++cc) {
            float e = __expf(SL[wave][row][cb + cc] - mx);
            ev[cc] = e;
            sum += e;
        }
        sum += __shfl_xor(sum, 1, 32);
        float inv = 1.0f / sum;
        #pragma unroll
        for (int cc = 0; cc < 32; ++cc)
            PL[wave][row][cb + cc] = (bf16_t)(ev[cc] * inv);
    }
    __syncthreads();

    // --- O = P * V  (16x32 strip, 2 K-chunks x 2 d-tiles = 4 WMMAs) ---
    v8f o[2] = {};
    #pragma unroll
    for (int kk = 0; kk < 2; ++kk) {
        v16bf pa = ld_a_frag(&PL[wave][l15][kk * 32 + hi * 16]);
        #pragma unroll
        for (int dt = 0; dt < 2; ++dt) {
            int d = dt * 16 + l15;
            v16bf vb = ld_b_frag(vh + (size_t)d * NPAD + kk * 32 + hi * 16);
            o[dt] = wmma_bf16(pa, vb, o[dt]);
        }
    }

    #pragma unroll
    for (int dt = 0; dt < 2; ++dt) {
        #pragma unroll
        for (int r = 0; r < 8; ++r) {
            int row = wave * 16 + r + 8 * hi;
            if (row < NWIN) {
                int col = h * HD + dt * 16 + l15;
                abuf[((size_t)b * NWIN + row) * CDIM + col] = (bf16_t)o[dt][r];
            }
        }
    }
}

// ---------------------------------------------------------------------------
// Kernel 3: output projection.  out[m][n] = sum_k A[m][k]*Wp[n][k] + pb[n]
// M=200704, K=256, N=256.  Grid (3136, 4), block 128.
// ---------------------------------------------------------------------------
__global__ __launch_bounds__(128)
void wa_proj_gemm(const bf16_t* __restrict__ ab, const bf16_t* __restrict__ wb,
                  const float* __restrict__ pbias, float* __restrict__ out) {
    const int lane   = threadIdx.x & 31;
    const int wave   = threadIdx.x >> 5;
    const int m_base = blockIdx.x * 64 + wave * 16;
    const int n_base = blockIdx.y * 64;
    const int l15    = lane & 15;
    const int hi     = lane >> 4;

    const bf16_t* aptr = ab + (size_t)(m_base + l15) * CDIM + hi * 16;
    const bf16_t* bptr = wb + (size_t)(n_base + l15) * CDIM + hi * 16;

    v8f c[4] = {};
    #pragma unroll
    for (int kk = 0; kk < 8; ++kk) {
        v16bf a = ld_a_frag(aptr + kk * 32);
        #pragma unroll
        for (int j = 0; j < 4; ++j) {
            v16bf b = ld_b_frag(bptr + (size_t)j * 16 * CDIM + kk * 32);
            c[j] = wmma_bf16(a, b, c[j]);
        }
    }

    #pragma unroll
    for (int j = 0; j < 4; ++j) {
        #pragma unroll
        for (int r = 0; r < 8; ++r) {
            int gm = m_base + r + 8 * hi;
            int n  = n_base + j * 16 + l15;
            out[(size_t)gm * CDIM + n] = c[j][r] + pbias[n];
        }
    }
}

// ---------------------------------------------------------------------------
extern "C" void kernel_launch(void* const* d_in, const int* in_sizes, int n_in,
                              void* d_out, int out_size, void* d_ws, size_t ws_size,
                              hipStream_t stream) {
    (void)in_sizes; (void)n_in; (void)out_size; (void)ws_size;

    const float* x      = (const float*)d_in[0];
    const float* mask   = (const float*)d_in[1];
    const float* rpb    = (const float*)d_in[2];
    const float* qkv_w  = (const float*)d_in[3];
    const float* qkv_b  = (const float*)d_in[4];
    const float* proj_w = (const float*)d_in[5];
    const float* proj_b = (const float*)d_in[6];
    const int*   rpi    = (const int*)d_in[7];
    float*       out    = (float*)d_out;

    // workspace carve-out (256B aligned)
    char*  ws  = (char*)d_ws;
    size_t off = 0;
    auto carve = [&](size_t bytes) -> char* {
        off = (off + 255) & ~(size_t)255;
        char* p = ws + off;
        off += bytes;
        return p;
    };
    bf16_t* xb    = (bf16_t*)carve((size_t)MTOT * CDIM * 2);
    bf16_t* qwb   = (bf16_t*)carve((size_t)3 * CDIM * CDIM * 2);
    bf16_t* pwb   = (bf16_t*)carve((size_t)CDIM * CDIM * 2);
    bf16_t* qbuf  = (bf16_t*)carve((size_t)4096 * NHD * NWIN * HD * 2);
    bf16_t* kbuf  = (bf16_t*)carve((size_t)4096 * NHD * NWIN * HD * 2);
    bf16_t* vtbuf = (bf16_t*)carve((size_t)4096 * NHD * HD * NPAD * 2);
    bf16_t* abuf  = (bf16_t*)carve((size_t)MTOT * CDIM * 2);

    // 0) convert inputs to bf16
    wa_cvt_bf16<<<8192, 256, 0, stream>>>(x, xb, (MTOT * CDIM) / 4);
    wa_cvt_bf16<<<768,  256, 0, stream>>>(qkv_w, qwb, (3 * CDIM * CDIM) / 4);
    wa_cvt_bf16<<<256,  256, 0, stream>>>(proj_w, pwb, (CDIM * CDIM) / 4);

    // 1) QKV projection
    wa_qkv_gemm<<<dim3(MTOT / 64, 12), 128, 0, stream>>>(
        xb, qwb, qkv_b, qbuf, kbuf, vtbuf);

    // 2) windowed attention
    wa_attn<<<dim3(4096, NHD), 128, 0, stream>>>(
        qbuf, kbuf, vtbuf, mask, rpb, rpi, abuf);

    // 3) output projection
    wa_proj_gemm<<<dim3(MTOT / 64, 4), 128, 0, stream>>>(
        abuf, pwb, proj_b, out);
}
